// Gaussian3D_76192719831188
// MI455X (gfx1250) — matrix operand order repaired and checked by reading, not compiled, not run on hardware
//
#include <hip/hip_runtime.h>

typedef __attribute__((ext_vector_type(16))) _Float16 v16h;
typedef __attribute__((ext_vector_type(8)))  _Float16 v8h;
typedef __attribute__((ext_vector_type(8)))  float    v8f;

#define N_ATOMS   2048
#define N_PIX     128
#define KCHUNK    32                    // K (atoms) per WMMA step
#define NCHUNKS   (N_ATOMS / KCHUNK)    // 64
#define XROWS     64                    // x rows staged per block (x-half)
#define XSTRIDE_H 40                    // LDS row stride in halfs (80 B, padded
                                        //  from 64 B -> conflict-free ds_load_b128)
#define BUF_H     (XROWS * XSTRIDE_H)   // 2560 halfs = 5120 B per buffer

// ---------------------------------------------------------------------------
// Kernel 1: per-axis voxel-averaged gaussian factor tables (erf integrals).
// All three tables stored f16, K-major ([p][a]) so WMMA fragments are
// contiguous 16 B runs:
//   gxB[x][a] : B-matrix table      gyH[y][a], gzH[z][a] : A-side factors
//   (amplitude folded into gz, matching reference gz*amp)
// ---------------------------------------------------------------------------
__global__ __launch_bounds__(256)
void Gaussian3D_axis_tables(const float* __restrict__ pos,        // [A,3]
                            const float* __restrict__ log_var,    // [1]
                            const float* __restrict__ log_weight, // [1]
                            _Float16* __restrict__ gyH,
                            _Float16* __restrict__ gzH,
                            _Float16* __restrict__ gxB,
                            const int* __restrict__ n_pix_p,
                            const float* __restrict__ voxel_p) {
  int t = blockIdx.x * blockDim.x + threadIdx.x;
  const int A = N_ATOMS;
  if (t >= 3 * N_PIX * A) return;
  int a    = t & (A - 1);
  int p    = (t / A) & (N_PIX - 1);
  int axis = t / (A * N_PIX);

  int   np    = *n_pix_p;
  float vs    = *voxel_p;
  float sigma = __expf(0.5f * log_var[0]);
  float amp   = __expf(log_weight[0]);
  float inv_denom = 1.0f / (1.41421356237f * sigma);
  float coord = (float)(p - (np >> 1)) * vs;
  float delta = coord - pos[a * 3 + axis];
  float h     = 0.5f * vs;
  float val = 0.5f * (erff((delta + h) * inv_denom) -
                      erff((delta - h) * inv_denom)) / vs;

  if (axis == 0)      gxB[p * A + a] = (_Float16)val;          // x factor
  else if (axis == 1) gyH[p * A + a] = (_Float16)val;          // y factor
  else                gzH[p * A + a] = (_Float16)(val * amp);  // z * amp
}

// ---------------------------------------------------------------------------
// Kernel 2: grid[z0, y, x] = sum_a (gz[a,z0]*gy[a,y]) * gx[a,x] as WMMA GEMM.
// Block = (z-slab, x-half): 256 blocks x 8 waves. Wave = one 16-row y tile
// x four 16-wide x tiles (one A-frag build feeds 4 WMMAs). The 4 KB gx K-panel
// is async-staged to LDS once per block (double buffered) and shared by all
// 8 waves. Per chunk: 4 global b128 + 8 pk_mul_f16 (A), 8 ds b128 (B),
// 4 WMMAs, 1 async issue, 2 barriers. This form keeps the accumulators
// pinned in their home VGPRs across the whole K loop (no copy chains);
// the per-tile dscnt waits are hidden by co-resident waves.
// ---------------------------------------------------------------------------
__global__ __launch_bounds__(256)
void Gaussian3D_render_wmma(const _Float16* __restrict__ gyH,
                            const _Float16* __restrict__ gzH,
                            const _Float16* __restrict__ gxB,
                            float* __restrict__ grid) {
  __shared__ __align__(16) _Float16 lds_gx[2 * BUF_H];

  const int A = N_ATOMS, P = N_PIX;
  const int z0    = blockIdx.x >> 1;
  const int xbase = (blockIdx.x & 1) << 6;   // 0 or 64
  const int wave  = threadIdx.x >> 5;        // 0..7 -> y tile
  const int lane  = threadIdx.x & 31;
  const int y0    = wave << 4;
  const int m     = lane & 15;
  const int hi    = lane >> 4;

  // Async staging assignment: 256 threads x 16 B = 4 KB per K chunk.
  const int sx = threadIdx.x >> 2;           // x row within the half (0..63)
  const int sq = threadIdx.x & 3;            // 16 B quarter of the 64 B row
  const _Float16* gsrc = gxB + (size_t)(xbase + sx) * A + sq * 8;
  const unsigned  ldst =
      (unsigned)(uintptr_t)(&lds_gx[0]) + sx * 80 + sq * 16;  // byte offset

  const _Float16* gyrow = gyH + (size_t)(y0 + m) * A;  // this lane's W row
  const _Float16* gzrow = gzH + (size_t)z0 * A;        // z factor row (shared)

  v8f c[4];
#pragma unroll
  for (int nt = 0; nt < 4; ++nt) c[nt] = (v8f){};

#define ISSUE(chunk)                                                        \
  asm volatile("global_load_async_to_lds_b128 %0, %1, off"                  \
               :: "v"(ldst + ((chunk) & 1) * (BUF_H * 2)),                  \
                  "v"(gsrc + (size_t)(chunk) * KCHUNK) : "memory")

  ISSUE(0);

  for (int i = 0; i < NCHUNKS; ++i) {
    if (i + 1 < NCHUNKS) {
      ISSUE(i + 1);
      // chunk i+1 in flight; in-order completion -> cnt<=1 means chunk i done
      asm volatile("s_wait_asynccnt 0x1" ::: "memory");
    } else {
      asm volatile("s_wait_asynccnt 0x0" ::: "memory");
    }
    __syncthreads();  // chunk i visible to all waves

    // --- A fragment (16-bit A 16x32 layout): W[m,k] = gy[a,y0+m]*gz[a,z0] ---
    // lanes 0-15: K = 0..7 (V0-3), 16..23 (V4-7); lanes 16-31: +8
    int a0    = i * KCHUNK;
    int base1 = a0 + hi * 8;        // halves 0..7
    int base2 = a0 + 16 + hi * 8;   // halves 8..15
    v8h gy_lo = *reinterpret_cast<const v8h*>(gyrow + base1);
    v8h gy_hi = *reinterpret_cast<const v8h*>(gyrow + base2);
    v8h gz_lo = *reinterpret_cast<const v8h*>(gzrow + base1);
    v8h gz_hi = *reinterpret_cast<const v8h*>(gzrow + base2);
    v8h w_lo = gy_lo * gz_lo;       // v_pk_mul_f16 x4
    v8h w_hi = gy_hi * gz_hi;       // v_pk_mul_f16 x4
    v16h afrag = __builtin_shufflevector(
        w_lo, w_hi, 0, 1, 2, 3, 4, 5, 6, 7, 8, 9, 10, 11, 12, 13, 14, 15);

    // --- 4 B fragments from LDS (B 32x16: same K order as A: lanes 0-15 K=0..7,16..23; 16-31 +8)
    const _Float16* lbuf = &lds_gx[(i & 1) * BUF_H];
#pragma unroll
    for (int nt = 0; nt < 4; ++nt) {
      const _Float16* lp = lbuf + (nt * 16 + m) * XSTRIDE_H + hi * 8;
      v8h blo = *reinterpret_cast<const v8h*>(lp);
      v8h bhi = *reinterpret_cast<const v8h*>(lp + 16);
      v16h bfrag = __builtin_shufflevector(
          blo, bhi, 0, 1, 2, 3, 4, 5, 6, 7, 8, 9, 10, 11, 12, 13, 14, 15);
      c[nt] = __builtin_amdgcn_wmma_f32_16x16x32_f16(
          /*neg_a=*/false, afrag, /*neg_b=*/false, bfrag,
          /*c_mod=*/(short)0, c[nt], /*reuse_a=*/false, /*reuse_b=*/false);
    }
    __syncthreads();  // all waves done with buf (i&1) before it is re-filled
  }
#undef ISSUE

  // --- C/D store: lanes 0-15 hold M = r, lanes 16-31 hold M = 8 + r --------
#pragma unroll
  for (int nt = 0; nt < 4; ++nt) {
    int x = xbase + nt * 16 + m;
#pragma unroll
    for (int r = 0; r < 8; ++r) {
      int row = y0 + hi * 8 + r;
      grid[((size_t)z0 * P + row) * P + x] = c[nt][r];
    }
  }
}

// ---------------------------------------------------------------------------
extern "C" void kernel_launch(void* const* d_in, const int* in_sizes, int n_in,
                              void* d_out, int out_size, void* d_ws, size_t ws_size,
                              hipStream_t stream) {
  const float* pos        = (const float*)d_in[0];  // [2048,3] f32
  const float* log_var    = (const float*)d_in[1];  // [1] f32
  const float* log_weight = (const float*)d_in[2];  // [1] f32
  const int*   n_pix_p    = (const int*)d_in[3];    // scalar int
  const float* voxel_p    = (const float*)d_in[4];  // scalar f32
  float* grid = (float*)d_out;                      // [128,128,128] f32

  // Workspace carve-up (1.5 MB): gyH | gzH | gxB, all f16 [P][A]
  _Float16* gyH = (_Float16*)d_ws;
  _Float16* gzH = gyH + (size_t)N_PIX * N_ATOMS;
  _Float16* gxB = gzH + (size_t)N_PIX * N_ATOMS;

  int total = 3 * N_PIX * N_ATOMS;
  hipLaunchKernelGGL(Gaussian3D_axis_tables, dim3((total + 255) / 256), dim3(256),
                     0, stream, pos, log_var, log_weight, gyH, gzH, gxB,
                     n_pix_p, voxel_p);

  // 128 z-slabs x 2 x-halves = 256 blocks; 8 wave32 per block.
  hipLaunchKernelGGL(Gaussian3D_render_wmma, dim3(256), dim3(256),
                     0, stream, gyH, gzH, gxB, grid);
}
